// SemiSupervisedRBM_48438641164625
// MI455X (gfx1250) — matrix operand order repaired and checked by reading, not compiled, not run on hardware
//
#include <hip/hip_runtime.h>

// ---------------------------------------------------------------------------
// SemiSupervisedRBM unlabeled loss (CD-10) for MI455X (gfx1250).
// Fully fused persistent-tile design: one workgroup owns 32 batch rows and
// runs the entire Gibbs chain in LDS (v tile 32x4096 bf16 ~= 257 KB exploits
// the 320 KB WGP LDS). v/h never touch HBM; inputs stream exactly once.
// Each loaded W B-fragment is multiplied against BOTH 16-row halves, halving
// the dominant L2 traffic on W. All matmuls via v_wmma_f32_16x16x32_bf16.
// ---------------------------------------------------------------------------

typedef __attribute__((ext_vector_type(16))) __bf16 v16bf;
typedef __attribute__((ext_vector_type(8)))  __bf16 v8bf;
typedef __attribute__((ext_vector_type(8)))  float  v8f;
typedef __attribute__((ext_vector_type(4)))  float  v4f;

#define B_ROWS    8192
#define V_DIM     4096
#define H_DIM     64
#define CD_STEPS  10
#define R_PER_BLK 32
#define VT_STRIDE 4104   // padded v-tile row stride (elems): 8208 B -> 4-bank row skew
#define HT_STRIDE 72     // padded h-tile row stride (elems): 144 B -> 36-bank row skew

// --- deterministic per-element hash RNG (stateless; same every launch) ------
__device__ __forceinline__ float urand(unsigned idx, unsigned salt) {
    unsigned x = idx * 0x9E3779B9u + salt * 0x85EBCA6Bu + 0x27220A95u;
    x ^= x >> 16; x *= 0x7FEB352Du;
    x ^= x >> 15; x *= 0x846CA68Bu;
    x ^= x >> 16;
    return (float)(x >> 8) * (1.0f / 16777216.0f);
}

__device__ __forceinline__ float sigmoidf(float x) {
    return 1.0f / (1.0f + __expf(-x));
}
__device__ __forceinline__ float softplusf(float x) {
    return fmaxf(x, 0.0f) + log1pf(__expf(-fabsf(x)));
}

__device__ __forceinline__ v8f wmma_bf16(v16bf a, v16bf b, v8f c) {
    return __builtin_amdgcn_wmma_f32_16x16x32_bf16(false, a, false, b,
                                                   (short)0, c, false, false);
}

// ---------------------------------------------------------------------------
__global__ void zero_out_kernel(float* out) { out[0] = 0.0f; }

// ---------------------------------------------------------------------------
// Pack W once per launch:
//  wrow : bf16 copy of W, row-major [V_DIM, H_DIM] (B operand of h @ W^T:
//         B[k,n] = W[n,k] -> 16 contiguous bf16 per lane)
//  wpack: W pre-swizzled into WMMA B-fragment order for v @ W:
//         wpack[((t*4+nt)*32+lane)*16+i] = W[t*32+(i>>3)*16+(lane>>4)*8+(i&7), nt*16+(lane&15)]
// ---------------------------------------------------------------------------
__global__ void pack_w_kernel(const float* __restrict__ w,
                              __bf16* __restrict__ wrow,
                              __bf16* __restrict__ wpack) {
    int idx = blockIdx.x * blockDim.x + threadIdx.x;
    if (idx >= V_DIM * H_DIM) return;
    wrow[idx] = (__bf16)w[idx];
    int i    = idx & 15;
    int lane = (idx >> 4) & 31;
    int nt   = (idx >> 9) & 3;
    int t    = idx >> 11;                       // 0..127
    int k    = t * 32 + (i >> 3) * 16 + (lane >> 4) * 8 + (i & 7);   // 0..4095
    int n    = nt * 16 + (lane & 15);           // 0..63
    wpack[idx] = (__bf16)w[k * H_DIM + n];
}

// ---------------------------------------------------------------------------
// Persistent fused CD-k chain + free-energy gap. Grid = 256 blocks x 256 thr.
// ---------------------------------------------------------------------------
__global__ __launch_bounds__(256)
void rbm_chain_kernel(const float* __restrict__ x,
                      const __bf16* __restrict__ wrow,
                      const __bf16* __restrict__ wpack,
                      const float* __restrict__ vbias,
                      const float* __restrict__ hbias,
                      float* __restrict__ out) {
    __shared__ __bf16 vtile[R_PER_BLK * VT_STRIDE];   // 262,656 B
    __shared__ __bf16 htile[R_PER_BLK * HT_STRIDE];   //   4,608 B
    __shared__ float  ltile[R_PER_BLK * H_DIM];       //   8,192 B

    const int tid   = threadIdx.x;
    const int lane  = tid & 31;
    const int wave  = tid >> 5;        // 0..7
    const int hs    = lane >> 4;       // half-wave select
    const int mloc  = lane & 15;       // local row within a 16-row half
    const int mBase = blockIdx.x * R_PER_BLK;
    const v8f vzero = {0.f, 0.f, 0.f, 0.f, 0.f, 0.f, 0.f, 0.f};

    float fe = 0.0f;   // this thread's contribution to sum(FE(x) - FE(v))

    for (int i = tid; i < R_PER_BLK * H_DIM; i += 256) ltile[i] = 0.0f;
    __syncthreads();

    // ============ init: logits = x @ W (K-split 512/wave, 2 row-halves) ====
    // Also folds in FE(x): -(x.vb) here, -softplus in the epilogue.
    {
        const float* Af0 = x + (size_t)(mBase + mloc) * V_DIM;
        const float* Af1 = x + (size_t)(mBase + 16 + mloc) * V_DIM;
        const int kBase = wave * 512;
        v8f acc[2][4];
#pragma unroll
        for (int mt = 0; mt < 2; ++mt)
#pragma unroll
            for (int nt = 0; nt < 4; ++nt) acc[mt][nt] = vzero;
        float bx = 0.0f;
        for (int kt = 0; kt < 16; ++kt) {
            const int k0 = kBase + kt * 32;
            const int kA = k0 + 8 * hs;
            const int kB = k0 + 16 + 8 * hs;
            v4f c0 = *(const v4f*)(vbias + kA);
            v4f c1 = *(const v4f*)(vbias + kA + 4);
            v4f c2 = *(const v4f*)(vbias + kB);
            v4f c3 = *(const v4f*)(vbias + kB + 4);
            v16bf a0, a1;
            {
                v4f p0 = *(const v4f*)(Af0 + kA);
                v4f p1 = *(const v4f*)(Af0 + kA + 4);
                v4f p2 = *(const v4f*)(Af0 + kB);
                v4f p3 = *(const v4f*)(Af0 + kB + 4);
                v4f q0 = *(const v4f*)(Af1 + kA);
                v4f q1 = *(const v4f*)(Af1 + kA + 4);
                v4f q2 = *(const v4f*)(Af1 + kB);
                v4f q3 = *(const v4f*)(Af1 + kB + 4);
#pragma unroll
                for (int i = 0; i < 4; ++i) {
                    a0[i]      = (__bf16)p0[i];
                    a0[4 + i]  = (__bf16)p1[i];
                    a0[8 + i]  = (__bf16)p2[i];
                    a0[12 + i] = (__bf16)p3[i];
                    a1[i]      = (__bf16)q0[i];
                    a1[4 + i]  = (__bf16)q1[i];
                    a1[8 + i]  = (__bf16)q2[i];
                    a1[12 + i] = (__bf16)q3[i];
                    bx += p0[i] * c0[i] + p1[i] * c1[i] + p2[i] * c2[i] + p3[i] * c3[i];
                    bx += q0[i] * c0[i] + q1[i] * c1[i] + q2[i] * c2[i] + q3[i] * c3[i];
                }
            }
            const size_t t = (size_t)(k0 >> 5);
#pragma unroll
            for (int nt = 0; nt < 4; ++nt) {
                v16bf b = *(const v16bf*)(wpack + (((t * 4 + nt) * 32 + lane) << 4));
                acc[0][nt] = wmma_bf16(a0, b, acc[0][nt]);
                acc[1][nt] = wmma_bf16(a1, b, acc[1][nt]);
            }
        }
        fe -= bx;  // -(x . vb)
#pragma unroll
        for (int mt = 0; mt < 2; ++mt)
#pragma unroll
            for (int nt = 0; nt < 4; ++nt)
#pragma unroll
                for (int r = 0; r < 8; ++r)
                    atomicAdd(&ltile[(mt * 16 + r + 8 * hs) * H_DIM + nt * 16 + mloc],
                              acc[mt][nt][r]);
    }
    __syncthreads();
    // init epilogue: h0 sample + FE(x) softplus
    for (int i = tid; i < R_PER_BLK * H_DIM; i += 256) {
        const int r = i >> 6, n = i & 63;
        const float logit = ltile[i] + hbias[n];
        fe -= softplusf(logit);
        const float u = urand((unsigned)((mBase + r) * H_DIM + n), 1u);
        htile[r * HT_STRIDE + n] = (sigmoidf(logit) > u) ? (__bf16)1.0f : (__bf16)0.0f;
    }
    __syncthreads();

    // ============ CD-k Gibbs chain, entirely in LDS ========================
    for (int s = 0; s < CD_STEPS; ++s) {
        const bool last = (s == CD_STEPS - 1);

        // reset hidden-logit accumulator (ltile idle since last barrier)
        for (int i = tid; i < R_PER_BLK * H_DIM; i += 256) ltile[i] = 0.0f;

        // ---- visible phase: v = sample(sigmoid(h @ W^T + vb)) ----
        {
            v16bf ah[2][2];   // [row-half][k-half]
#pragma unroll
            for (int mt = 0; mt < 2; ++mt) {
                const __bf16* Ah = htile + (mt * 16 + mloc) * HT_STRIDE;
                v8bf q0 = *(const v8bf*)(Ah + 16 * hs);
                v8bf q1 = *(const v8bf*)(Ah + 8 + 16 * hs);
                v8bf q2 = *(const v8bf*)(Ah + 32 + 16 * hs);
                v8bf q3 = *(const v8bf*)(Ah + 40 + 16 * hs);
#pragma unroll
                for (int i = 0; i < 8; ++i) {
                    ah[mt][0][i] = q0[i]; ah[mt][0][8 + i] = q1[i];
                    ah[mt][1][i] = q2[i]; ah[mt][1][8 + i] = q3[i];
                }
            }
            for (int cg = 0; cg < 8; ++cg) {
                const int nBase = (wave * 8 + cg) * 64;
                v8f acc[2][4];
#pragma unroll
                for (int mt = 0; mt < 2; ++mt)
#pragma unroll
                    for (int nt = 0; nt < 4; ++nt) acc[mt][nt] = vzero;
#pragma unroll
                for (int half = 0; half < 2; ++half) {
                    const int kh = half * 32 + hs * 16;
#pragma unroll
                    for (int nt = 0; nt < 4; ++nt) {
                        const int n = nBase + nt * 16 + mloc;
                        v16bf b = *(const v16bf*)(wrow + (size_t)n * H_DIM + kh);
                        acc[0][nt] = wmma_bf16(ah[0][half], b, acc[0][nt]);
                        acc[1][nt] = wmma_bf16(ah[1][half], b, acc[1][nt]);
                    }
                }
#pragma unroll
                for (int mt = 0; mt < 2; ++mt)
#pragma unroll
                    for (int nt = 0; nt < 4; ++nt) {
                        const int n = nBase + nt * 16 + mloc;
                        const float bias = vbias[n];
#pragma unroll
                        for (int r = 0; r < 8; ++r) {
                            const int rl = mt * 16 + r + 8 * hs;
                            const float p = sigmoidf(acc[mt][nt][r] + bias);
                            const float u =
                                urand((unsigned)((mBase + rl) * V_DIM + n) * 2654435761u,
                                      100u + 2u * s);
                            vtile[rl * VT_STRIDE + n] = (p > u) ? (__bf16)1.0f : (__bf16)0.0f;
                        }
                    }
            }
        }
        __syncthreads();

        // ---- hidden phase: logits = v @ W (K-split 512/wave, 2 row-halves)
        // On the last step also fold in +(v . vb) for FE(v).
        {
            const __bf16* Av0 = vtile + (size_t)mloc * VT_STRIDE;
            const __bf16* Av1 = vtile + (size_t)(16 + mloc) * VT_STRIDE;
            const int kBase = wave * 512;
            v8f acc[2][4];
#pragma unroll
            for (int mt = 0; mt < 2; ++mt)
#pragma unroll
                for (int nt = 0; nt < 4; ++nt) acc[mt][nt] = vzero;
            float bv = 0.0f;
            for (int kt = 0; kt < 16; ++kt) {
                const int k0 = kBase + kt * 32;
                const int kA = k0 + 8 * hs;
                const int kB = k0 + 16 + 8 * hs;
                v8bf q00 = *(const v8bf*)(Av0 + kA);
                v8bf q01 = *(const v8bf*)(Av0 + kB);
                v8bf q10 = *(const v8bf*)(Av1 + kA);
                v8bf q11 = *(const v8bf*)(Av1 + kB);
                v16bf a0, a1;
#pragma unroll
                for (int i = 0; i < 8; ++i) {
                    a0[i] = q00[i]; a0[8 + i] = q01[i];
                    a1[i] = q10[i]; a1[8 + i] = q11[i];
                }
                if (last) {
                    v4f c0 = *(const v4f*)(vbias + kA);
                    v4f c1 = *(const v4f*)(vbias + kA + 4);
                    v4f c2 = *(const v4f*)(vbias + kB);
                    v4f c3 = *(const v4f*)(vbias + kB + 4);
#pragma unroll
                    for (int i = 0; i < 4; ++i) {
                        bv += (float)q00[i] * c0[i] + (float)q00[4 + i] * c1[i] +
                              (float)q01[i] * c2[i] + (float)q01[4 + i] * c3[i];
                        bv += (float)q10[i] * c0[i] + (float)q10[4 + i] * c1[i] +
                              (float)q11[i] * c2[i] + (float)q11[4 + i] * c3[i];
                    }
                }
                const size_t t = (size_t)(k0 >> 5);
#pragma unroll
                for (int nt = 0; nt < 4; ++nt) {
                    v16bf b = *(const v16bf*)(wpack + (((t * 4 + nt) * 32 + lane) << 4));
                    acc[0][nt] = wmma_bf16(a0, b, acc[0][nt]);
                    acc[1][nt] = wmma_bf16(a1, b, acc[1][nt]);
                }
            }
            if (last) fe += bv;  // +(v . vb)
#pragma unroll
            for (int mt = 0; mt < 2; ++mt)
#pragma unroll
                for (int nt = 0; nt < 4; ++nt)
#pragma unroll
                    for (int r = 0; r < 8; ++r)
                        atomicAdd(&ltile[(mt * 16 + r + 8 * hs) * H_DIM + nt * 16 + mloc],
                                  acc[mt][nt][r]);
        }
        __syncthreads();

        // ---- hidden epilogue: h sample (+ FE(v) softplus on last step) ----
        for (int i = tid; i < R_PER_BLK * H_DIM; i += 256) {
            const int r = i >> 6, n = i & 63;
            const float logit = ltile[i] + hbias[n];
            if (last) fe += softplusf(logit);  // + sum_j softplus(v W + hb)
            const float u = urand((unsigned)((mBase + r) * H_DIM + n), 101u + 2u * s);
            htile[r * HT_STRIDE + n] = (sigmoidf(logit) > u) ? (__bf16)1.0f : (__bf16)0.0f;
        }
        __syncthreads();
    }

    // ============ final reduction: wave shuffle + one atomic per wave ======
#pragma unroll
    for (int off = 1; off < 32; off <<= 1) fe += __shfl_xor(fe, off, 32);
    if (lane == 0) atomicAdd(out, fe);
}

// ---------------------------------------------------------------------------
// Host-side orchestration (graph-capture safe: stream-ordered launches only)
// ---------------------------------------------------------------------------
extern "C" void kernel_launch(void* const* d_in, const int* in_sizes, int n_in,
                              void* d_out, int out_size, void* d_ws, size_t ws_size,
                              hipStream_t stream) {
    const float* inputs = (const float*)d_in[0];   // [8192, 4096]
    const float* w      = (const float*)d_in[1];   // [4096, 64]
    const float* vb     = (const float*)d_in[2];   // [4096]
    const float* hb     = (const float*)d_in[3];   // [64]
    float* out = (float*)d_out;

    char* ws = (char*)d_ws;
    __bf16* w_row  = (__bf16*)ws;                                    // 512 KiB
    __bf16* w_pack = (__bf16*)(ws + (size_t)V_DIM * H_DIM * 2);      // 512 KiB

    zero_out_kernel<<<1, 1, 0, stream>>>(out);
    pack_w_kernel<<<(V_DIM * H_DIM + 255) / 256, 256, 0, stream>>>(w, w_row, w_pack);
    rbm_chain_kernel<<<B_ROWS / R_PER_BLK, 256, 0, stream>>>(inputs, w_row, w_pack,
                                                             vb, hb, out);
}
